// CustomAttnDecoderRNN_17008070492416
// MI455X (gfx1250) — matrix operand order repaired and checked by reading, not compiled, not run on hardware
//
#include <hip/hip_runtime.h>
#include <hip/hip_bf16.h>
#include <math.h>

// ---------------------------------------------------------------------------
// Shapes (compile-time constants from the reference)
// ---------------------------------------------------------------------------
#define Bsz 64
#define Ssz 128
#define Esz 512
#define Hsz 1024
#define Vsz 32000
#define G3  (3 * Hsz)          // 3072

typedef __attribute__((ext_vector_type(16))) __bf16 v16bf;
typedef __attribute__((ext_vector_type(8)))  __bf16 v8bf;
typedef __attribute__((ext_vector_type(4)))  __bf16 v4bf;
typedef __attribute__((ext_vector_type(8)))  float  v8f;
typedef __attribute__((ext_vector_type(4)))  float  f4v;

#if __has_builtin(__builtin_amdgcn_tanhf)
#define fast_tanh(x) __builtin_amdgcn_tanhf(x)
#else
#define fast_tanh(x) tanhf(x)
#endif

// ---------------------------------------------------------------------------
// WMMA fragment layout (CDNA5 wave32, v_wmma_f32_16x16x32_bf16)
//
// A fragment (16x32, bf16): lane m = lane&15; half = lane>>4.
//   elements 0..7  -> K = k0 + half*16 + e
//   elements 8..15 -> K = k0 + half*16 + e
// B fragment (32x16, bf16): lane n = lane&15; half = lane>>4.
//   elements 0..15 -> K = k0 + half*16 + e
// C/D (16x16 f32, 8 VGPRs): lane holds N=lane&15; reg r -> M = r + (lane>>4)*8
// ---------------------------------------------------------------------------

// fp32 source + in-register convert (for single-use / tiny streams)
__device__ __forceinline__ v16bf load_a_frag(const float* A, int lda, int m0, int k0) {
    const int lane = threadIdx.x & 31;
    const float* p = A + (size_t)(m0 + (lane & 15)) * lda + k0 + ((lane >> 4) << 4);
    f4v x0 = *(const f4v*)(p);
    f4v x1 = *(const f4v*)(p + 4);
    f4v x2 = *(const f4v*)(p + 8);
    f4v x3 = *(const f4v*)(p + 12);
    v16bf a;
    a[0]  = (__bf16)x0.x; a[1]  = (__bf16)x0.y; a[2]  = (__bf16)x0.z; a[3]  = (__bf16)x0.w;
    a[4]  = (__bf16)x1.x; a[5]  = (__bf16)x1.y; a[6]  = (__bf16)x1.z; a[7]  = (__bf16)x1.w;
    a[8]  = (__bf16)x2.x; a[9]  = (__bf16)x2.y; a[10] = (__bf16)x2.z; a[11] = (__bf16)x2.w;
    a[12] = (__bf16)x3.x; a[13] = (__bf16)x3.y; a[14] = (__bf16)x3.z; a[15] = (__bf16)x3.w;
    return a;
}

__device__ __forceinline__ v16bf load_b_frag(const float* W, int ldw, int n0, int k0) {
    const int lane = threadIdx.x & 31;
    const float* p = W + (size_t)(n0 + (lane & 15)) * ldw + k0 + ((lane >> 4) << 4);
    f4v x0 = *(const f4v*)(p);
    f4v x1 = *(const f4v*)(p + 4);
    f4v x2 = *(const f4v*)(p + 8);
    f4v x3 = *(const f4v*)(p + 12);
    v16bf b;
    b[0]  = (__bf16)x0.x; b[1]  = (__bf16)x0.y; b[2]  = (__bf16)x0.z; b[3]  = (__bf16)x0.w;
    b[4]  = (__bf16)x1.x; b[5]  = (__bf16)x1.y; b[6]  = (__bf16)x1.z; b[7]  = (__bf16)x1.w;
    b[8]  = (__bf16)x2.x; b[9]  = (__bf16)x2.y; b[10] = (__bf16)x2.z; b[11] = (__bf16)x2.w;
    b[12] = (__bf16)x3.x; b[13] = (__bf16)x3.y; b[14] = (__bf16)x3.z; b[15] = (__bf16)x3.w;
    return b;
}

// fp32 source, NON-TEMPORAL: single-use weight streams (Wout 131MB, gates 44MB)
// must not evict the reused working set from the 192MB L2.
__device__ __forceinline__ v16bf load_b_frag_nt(const float* W, int ldw, int n0, int k0) {
    const int lane = threadIdx.x & 31;
    const float* p = W + (size_t)(n0 + (lane & 15)) * ldw + k0 + ((lane >> 4) << 4);
    f4v x0 = __builtin_nontemporal_load((const f4v*)(p));
    f4v x1 = __builtin_nontemporal_load((const f4v*)(p + 4));
    f4v x2 = __builtin_nontemporal_load((const f4v*)(p + 8));
    f4v x3 = __builtin_nontemporal_load((const f4v*)(p + 12));
    v16bf b;
    b[0]  = (__bf16)x0.x; b[1]  = (__bf16)x0.y; b[2]  = (__bf16)x0.z; b[3]  = (__bf16)x0.w;
    b[4]  = (__bf16)x1.x; b[5]  = (__bf16)x1.y; b[6]  = (__bf16)x1.z; b[7]  = (__bf16)x1.w;
    b[8]  = (__bf16)x2.x; b[9]  = (__bf16)x2.y; b[10] = (__bf16)x2.z; b[11] = (__bf16)x2.w;
    b[12] = (__bf16)x3.x; b[13] = (__bf16)x3.y; b[14] = (__bf16)x3.z; b[15] = (__bf16)x3.w;
    return b;
}

// pre-converted bf16 source (zero-cvt hot path for k_scores)
__device__ __forceinline__ v16bf load_a_frag_bf(const __bf16* A, int lda, int m0, int k0) {
    const int lane = threadIdx.x & 31;
    const __bf16* p = A + (size_t)(m0 + (lane & 15)) * lda + k0 + ((lane >> 4) << 4);
    v8bf lo = *(const v8bf*)(p);        // K = base .. base+7      (16B load)
    v8bf hi = *(const v8bf*)(p + 8);    // K = base+8 .. base+15   (16B load)
    v16bf a;
#pragma unroll
    for (int e = 0; e < 8; ++e) { a[e] = lo[e]; a[8 + e] = hi[e]; }
    return a;
}

__device__ __forceinline__ v16bf load_b_frag_bf(const __bf16* W, int ldw, int n0, int k0) {
    const int lane = threadIdx.x & 31;
    const __bf16* p = W + (size_t)(n0 + (lane & 15)) * ldw + k0 + ((lane >> 4) << 4);
    return *(const v16bf*)(p);          // 32B load -> 2x b128
}

__device__ __forceinline__ v8f wmma_bf16(v16bf a, v16bf b, v8f c) {
    return __builtin_amdgcn_wmma_f32_16x16x32_bf16(false, a, false, b, (short)0, c,
                                                   false, false);
}

__device__ __forceinline__ v8f vzero() {
    v8f z = {0.f, 0.f, 0.f, 0.f, 0.f, 0.f, 0.f, 0.f};
    return z;
}

// ---------------------------------------------------------------------------
// K0a: contiguous fp32 -> bf16 conversion (enc)
// ---------------------------------------------------------------------------
__global__ void k_cvt(const float* __restrict__ src, __bf16* __restrict__ dst, int n4) {
    const int i = blockIdx.x * 256 + threadIdx.x;
    if (i >= n4) return;
    const f4v x = *(const f4v*)(src + (size_t)i * 4);
    v4bf o;
    o[0] = (__bf16)x.x; o[1] = (__bf16)x.y; o[2] = (__bf16)x.z; o[3] = (__bf16)x.w;
    *(v4bf*)(dst + (size_t)i * 4) = o;
}

// K0b: repack Wa[:, H:] (stride 2H) into dense bf16 [H, H]
__global__ void k_cvt_wa2(const float* __restrict__ Wa, __bf16* __restrict__ dst) {
    const int i = blockIdx.x * 256 + threadIdx.x;   // over H*H/4
    const int row  = i >> 8;                        // H/4 = 256 float4 per row
    const int col4 = i & 255;
    const f4v x = *(const f4v*)(Wa + (size_t)row * 2 * Hsz + Hsz + col4 * 4);
    v4bf o;
    o[0] = (__bf16)x.x; o[1] = (__bf16)x.y; o[2] = (__bf16)x.z; o[3] = (__bf16)x.w;
    *(v4bf*)(dst + (size_t)row * Hsz + col4 * 4) = o;
}

// ---------------------------------------------------------------------------
// K1: Pre[b,n] = (h @ Wa1.T)[b,n] + ba[n]        (Wa1 = Wa[:, :H], ldw = 2H)
// waves: 4 m-tiles x 64 n-tiles = 256
// ---------------------------------------------------------------------------
__global__ void k_pre(const float* __restrict__ h, const float* __restrict__ Wa,
                      const float* __restrict__ ba, float* __restrict__ Pre) {
    const int gw = (blockIdx.x * blockDim.x + threadIdx.x) >> 5;
    const int mt = gw >> 6;          // 0..3
    const int nt = gw & 63;          // 0..63
    if (mt >= 4) return;
    v8f acc = vzero();
#pragma unroll 4
    for (int kt = 0; kt < Hsz / 32; ++kt) {
        v16bf a = load_a_frag(h, Hsz, mt * 16, kt * 32);
        v16bf b = load_b_frag(Wa, 2 * Hsz, nt * 16, kt * 32);
        acc = wmma_bf16(a, b, acc);
    }
    const int lane  = threadIdx.x & 31;
    const int n     = nt * 16 + (lane & 15);
    const int mbase = mt * 16 + ((lane >> 4) << 3);
    const float bn  = ba[n];
#pragma unroll
    for (int r = 0; r < 8; ++r)
        Pre[(size_t)(mbase + r) * Hsz + n] = acc[r] + bn;
}

// ---------------------------------------------------------------------------
// K2: scores[b,s] = sum_n v[n] * tanh( (enc_row @ Wa2.T)[m,n] + Pre[b,n] )
// Pure-bf16 operands (pre-converted): zero cvt in the hot loop.
// Register-blocked: 8 accumulators (16x128 per pass), A reused for 8 WMMAs.
// waves: 512 m-tiles
// ---------------------------------------------------------------------------
#define NBLK 8
__global__ void k_scores(const __bf16* __restrict__ encB, const __bf16* __restrict__ Wa2B,
                         const float* __restrict__ Pre, const float* __restrict__ v,
                         float* __restrict__ scores) {
    const int mt = (blockIdx.x * blockDim.x + threadIdx.x) >> 5;
    if (mt >= (Ssz * Bsz) / 16) return;
    const int lane  = threadIdx.x & 31;
    const int mbase = mt * 16 + ((lane >> 4) << 3);

    float sacc[8];
#pragma unroll
    for (int r = 0; r < 8; ++r) sacc[r] = 0.f;

    for (int ntc = 0; ntc < (Hsz / 16) / NBLK; ++ntc) {      // 8 chunks
        v8f acc[NBLK];
#pragma unroll
        for (int j = 0; j < NBLK; ++j) acc[j] = vzero();

        for (int kt = 0; kt < Hsz / 32; ++kt) {              // 32 k-steps
            v16bf a = load_a_frag_bf(encB, Hsz, mt * 16, kt * 32);
#pragma unroll
            for (int j = 0; j < NBLK; ++j) {
                v16bf b = load_b_frag_bf(Wa2B, Hsz, (ntc * NBLK + j) * 16, kt * 32);
                acc[j] = wmma_bf16(a, b, acc[j]);
            }
        }
#pragma unroll
        for (int j = 0; j < NBLK; ++j) {
            const int n    = (ntc * NBLK + j) * 16 + (lane & 15);
            const float vn = v[n];
#pragma unroll
            for (int r = 0; r < 8; ++r) {
                const int m  = mbase + r;
                const int bb = m & (Bsz - 1);
                const float e = fast_tanh(acc[j][r] + Pre[(size_t)bb * Hsz + n]);
                sacc[r] += e * vn;
            }
        }
    }
    // reduce across the 16 lanes that hold distinct n
#pragma unroll
    for (int off = 1; off < 16; off <<= 1)
#pragma unroll
        for (int r = 0; r < 8; ++r) sacc[r] += __shfl_xor(sacc[r], off, 32);

    if ((lane & 15) == 0) {
#pragma unroll
        for (int r = 0; r < 8; ++r) {
            const int m  = mbase + r;
            const int s  = m >> 6;          // m / B
            const int bb = m & (Bsz - 1);   // m % B
            scores[bb * Ssz + s] = sacc[r];
        }
    }
}

// ---------------------------------------------------------------------------
// K3: softmax over S; writes weights to ws and to the attn output slot
// ---------------------------------------------------------------------------
__global__ void k_softmax(const float* __restrict__ scores, float* __restrict__ attnw,
                          float* __restrict__ out_attn) {
    __shared__ float sm[Ssz];
    const int b = blockIdx.x, t = threadIdx.x;
    const float x = scores[b * Ssz + t];
    sm[t] = x;
    __syncthreads();
    for (int off = Ssz / 2; off > 0; off >>= 1) {
        if (t < off) sm[t] = fmaxf(sm[t], sm[t + off]);
        __syncthreads();
    }
    const float mx = sm[0];
    __syncthreads();
    const float e = __expf(x - mx);
    sm[t] = e;
    __syncthreads();
    for (int off = Ssz / 2; off > 0; off >>= 1) {
        if (t < off) sm[t] += sm[t + off];
        __syncthreads();
    }
    const float w = e / sm[0];
    attnw[b * Ssz + t]    = w;
    out_attn[b * Ssz + t] = w;
}

// ---------------------------------------------------------------------------
// K4: context[b,h] = sum_s w[b,s] * enc[s,b,h]   (coalesced over h)
// ---------------------------------------------------------------------------
__global__ void k_context(const float* __restrict__ enc, const float* __restrict__ attnw,
                          float* __restrict__ ctx) {
    __shared__ float w[Ssz];
    const int b = blockIdx.x;
    const int h = blockIdx.y * 256 + threadIdx.x;
    if (threadIdx.x < Ssz) w[threadIdx.x] = attnw[b * Ssz + threadIdx.x];
    __syncthreads();
    float acc = 0.f;
#pragma unroll 8
    for (int s = 0; s < Ssz; ++s)
        acc += w[s] * enc[((size_t)s * Bsz + b) * Hsz + h];
    ctx[(size_t)b * Hsz + h] = acc;
}

// ---------------------------------------------------------------------------
// K5: gate GEMMs.  gA = emb@Wx.T + ctx@Wc.T + z@Wz.T (+biases), gH = h@Wh.T + bh
// Weight streams are single-use -> non-temporal loads.
// waves: 4 m-tiles x 192 n-tiles = 768
// ---------------------------------------------------------------------------
__global__ void k_gates(const float* __restrict__ emb, const float* __restrict__ h,
                        const float* __restrict__ ctx, const float* __restrict__ z,
                        const float* __restrict__ Wx, const float* __restrict__ bx,
                        const float* __restrict__ Wh, const float* __restrict__ bh,
                        const float* __restrict__ Wc, const float* __restrict__ bc,
                        const float* __restrict__ Wz, const float* __restrict__ bz,
                        float* __restrict__ gA, float* __restrict__ gH) {
    const int gw = (blockIdx.x * blockDim.x + threadIdx.x) >> 5;
    const int mt = gw / (G3 / 16);
    const int nt = gw % (G3 / 16);
    if (mt >= 4) return;

    v8f accA = vzero();
    v8f accH = vzero();
#pragma unroll 4
    for (int kt = 0; kt < Esz / 32; ++kt)
        accA = wmma_bf16(load_a_frag(emb, Esz, mt * 16, kt * 32),
                         load_b_frag_nt(Wx, Esz, nt * 16, kt * 32), accA);
#pragma unroll 4
    for (int kt = 0; kt < Hsz / 32; ++kt)
        accA = wmma_bf16(load_a_frag(ctx, Hsz, mt * 16, kt * 32),
                         load_b_frag_nt(Wc, Hsz, nt * 16, kt * 32), accA);
#pragma unroll 4
    for (int kt = 0; kt < Hsz / 32; ++kt)
        accA = wmma_bf16(load_a_frag(z, Hsz, mt * 16, kt * 32),
                         load_b_frag_nt(Wz, Hsz, nt * 16, kt * 32), accA);
#pragma unroll 4
    for (int kt = 0; kt < Hsz / 32; ++kt)
        accH = wmma_bf16(load_a_frag(h, Hsz, mt * 16, kt * 32),
                         load_b_frag_nt(Wh, Hsz, nt * 16, kt * 32), accH);

    const int lane  = threadIdx.x & 31;
    const int n     = nt * 16 + (lane & 15);
    const int mbase = mt * 16 + ((lane >> 4) << 3);
    const float bA  = bx[n] + bc[n] + bz[n];
    const float bH  = bh[n];
#pragma unroll
    for (int r = 0; r < 8; ++r) {
        gA[(size_t)(mbase + r) * G3 + n] = accA[r] + bA;
        gH[(size_t)(mbase + r) * G3 + n] = accH[r] + bH;
    }
}

// ---------------------------------------------------------------------------
// K6: GRU elementwise combine -> h_new (ws + output slot 2)
// ---------------------------------------------------------------------------
__global__ void k_gru(const float* __restrict__ gA, const float* __restrict__ gH,
                      const float* __restrict__ h, float* __restrict__ hnew,
                      float* __restrict__ out_h) {
    const int i = blockIdx.x * 256 + threadIdx.x;   // i = b*H + j
    const int b = i >> 10, j = i & (Hsz - 1);
    const float* ga = gA + (size_t)b * G3;
    const float* gh = gH + (size_t)b * G3;
    const float r = 1.f / (1.f + __expf(-(ga[j] + gh[j])));
    const float u = 1.f / (1.f + __expf(-(ga[Hsz + j] + gh[Hsz + j])));
    const float n = fast_tanh(ga[2 * Hsz + j] + r * gh[2 * Hsz + j]);
    const float hv = h[i];
    const float hn = (1.f - u) * n + u * hv;
    hnew[i]  = hn;
    out_h[i] = hn;
}

// ---------------------------------------------------------------------------
// K7: logits = h_new @ Wout.T + bout.  HBM-bound on Wout (131 MB, single-use)
// -> non-temporal fp32 stream + in-register cvt.
// Register-blocked: 4 accumulators per wave (16x64 output).
// waves: 4 m-tiles x 500 n-groups = 2000
// ---------------------------------------------------------------------------
#define OBLK 4
__global__ void k_out(const float* __restrict__ hnew, const float* __restrict__ Wout,
                      const float* __restrict__ bout, float* __restrict__ out) {
    const int gw  = (blockIdx.x * blockDim.x + threadIdx.x) >> 5;
    const int mt  = gw / (Vsz / 16 / OBLK);
    const int ntg = gw % (Vsz / 16 / OBLK);
    if (mt >= 4) return;

    v8f acc[OBLK];
#pragma unroll
    for (int j = 0; j < OBLK; ++j) acc[j] = vzero();

    for (int kt = 0; kt < Hsz / 32; ++kt) {
        v16bf a = load_a_frag(hnew, Hsz, mt * 16, kt * 32);
#pragma unroll
        for (int j = 0; j < OBLK; ++j) {
            v16bf b = load_b_frag_nt(Wout, Hsz, (ntg * OBLK + j) * 16, kt * 32);
            acc[j] = wmma_bf16(a, b, acc[j]);
        }
    }
    const int lane  = threadIdx.x & 31;
    const int mbase = mt * 16 + ((lane >> 4) << 3);
#pragma unroll
    for (int j = 0; j < OBLK; ++j) {
        const int n    = (ntg * OBLK + j) * 16 + (lane & 15);
        const float bn = bout[n];
#pragma unroll
        for (int r = 0; r < 8; ++r)
            out[(size_t)(mbase + r) * Vsz + n] = acc[j][r] + bn;
    }
}

// ---------------------------------------------------------------------------
// Launch
// ---------------------------------------------------------------------------
extern "C" void kernel_launch(void* const* d_in, const int* in_sizes, int n_in,
                              void* d_out, int out_size, void* d_ws, size_t ws_size,
                              hipStream_t stream) {
    const float* emb  = (const float*)d_in[0];
    const float* hid  = (const float*)d_in[1];   // [1,B,H] -> h_last
    const float* enc  = (const float*)d_in[2];   // [S,B,H]
    const float* z    = (const float*)d_in[3];
    const float* Wa   = (const float*)d_in[4];
    const float* ba   = (const float*)d_in[5];
    const float* v    = (const float*)d_in[6];
    const float* Wx   = (const float*)d_in[7];
    const float* bx   = (const float*)d_in[8];
    const float* Wh   = (const float*)d_in[9];
    const float* bh   = (const float*)d_in[10];
    const float* Wc   = (const float*)d_in[11];
    const float* bc   = (const float*)d_in[12];
    const float* Wz   = (const float*)d_in[13];
    const float* bz   = (const float*)d_in[14];
    const float* Wout = (const float*)d_in[15];
    const float* bout = (const float*)d_in[16];

    float* out   = (float*)d_out;
    float* out_h = out + (size_t)Bsz * Vsz;                 // h_new slot
    float* out_a = out_h + (size_t)Bsz * Hsz;               // attn slot

    // workspace layout: bf16 regions first (256B-aligned base keeps vector
    // loads aligned), then fp32 scratch
    __bf16* wEncB = (__bf16*)d_ws;                          // S*B*H bf16 (16.8 MB)
    __bf16* wWa2B = wEncB + (size_t)Ssz * Bsz * Hsz;        // H*H   bf16 ( 2.1 MB)
    float*  wsf   = (float*)(wWa2B + (size_t)Hsz * Hsz);
    float* wPre    = wsf;                                   // 64*1024
    float* wScores = wPre + Bsz * Hsz;                      // 64*128
    float* wAttn   = wScores + Bsz * Ssz;                   // 64*128
    float* wCtx    = wAttn + Bsz * Ssz;                     // 64*1024
    float* wGA     = wCtx + Bsz * Hsz;                      // 64*3072
    float* wGH     = wGA + Bsz * G3;                        // 64*3072
    float* wHnew   = wGH + Bsz * G3;                        // 64*1024

    const int enc_n4 = (Ssz * Bsz * Hsz) / 4;               // 2,097,152
    k_cvt<<<enc_n4 / 256, 256, 0, stream>>>(enc, wEncB, enc_n4);
    k_cvt_wa2<<<(Hsz * Hsz / 4) / 256, 256, 0, stream>>>(Wa, wWa2B);

    k_pre<<<32, 256, 0, stream>>>(hid, Wa, ba, wPre);                       // 256 waves
    k_scores<<<64, 256, 0, stream>>>(wEncB, wWa2B, wPre, v, wScores);       // 512 waves
    k_softmax<<<Bsz, Ssz, 0, stream>>>(wScores, wAttn, out_a);
    k_context<<<dim3(Bsz, Hsz / 256), 256, 0, stream>>>(enc, wAttn, wCtx);
    k_gates<<<96, 256, 0, stream>>>(emb, hid, wCtx, z, Wx, bx, Wh, bh,
                                    Wc, bc, Wz, bz, wGA, wGH);              // 768 waves
    k_gru<<<(Bsz * Hsz) / 256, 256, 0, stream>>>(wGA, wGH, hid, wHnew, out_h);
    k_out<<<250, 256, 0, stream>>>(wHnew, Wout, bout, out);                 // 2000 waves
}
